// LinearAttention_61220463837901
// MI455X (gfx1250) — compile-verified, matrix operand order checked by reading, not run on hardware
//
#include <hip/hip_runtime.h>

// Linear attention (sigmoid feature map) for b=8, s=4096, d=256, fp32 in/out.
// HBM-bound (~128 MB min traffic @ 23.3 TB/s => ~5.5 us floor): read each
// tensor once, both GEMMs on v_wmma_f32_16x16x32_f16 (f32 accum), 256x256
// per-batch KV + k_sum in workspace (L2-resident). This revision: per-thread
// row-owned staging (b128 LDS stores, immediate-offset global clauses),
// double-buffered LDS in pass 1, register-prefetch pipeline in pass 2,
// k_sum folded into registers, normalizer folded into phi_Q staging.

typedef __attribute__((ext_vector_type(16))) _Float16     v16h;
typedef __attribute__((ext_vector_type(8)))  float        v8f;
typedef __attribute__((ext_vector_type(4)))  unsigned int v4u;

#define DFEAT  256
#define SLEN   4096
#define NBATCH 8
#define NCHUNK 16
#define SCHUNK (SLEN / NCHUNK)   // 256 s-rows per split-K chunk

__device__ __forceinline__ float phi_sig(float x) {
  // sigmoid(A*x - B) = 1 / (1 + exp(B - A*x))
  return 1.0f / (1.0f + __expf(4.102f - 0.6053f * x));
}

// Load one 16x32 f16 WMMA A/B fragment from an LDS tile stored row-major
// [row][k] (strideHalves halves per row). Fragment window = halves [k0,k0+32)
// of rows [r0, r0+16). Per CDNA5 ISA 16-bit layout: lanes 0-15 hold K=0..7 in
// v0..3 (2 halves/VGPR) and K=16..23 in v4..7; lanes 16-31 hold K=8..15 and
// K=24..31  ->  two 16-byte LDS loads per fragment.
__device__ __forceinline__ v16h load_frag16x32(const _Float16* base,
                                               int strideHalves, int r0, int k0) {
  const int lane = threadIdx.x & 31;
  const int row  = r0 + (lane & 15);
  const v4u* rp  = (const v4u*)(base + row * strideHalves + k0);
  const int sel  = (lane < 16) ? 0 : 1;
  union { v16h h; v4u u[2]; } f;
  f.u[0] = rp[sel];
  f.u[1] = rp[2 + sel];
  return f.h;
}

__global__ void la_zero_ws(float* __restrict__ ws, int n) {
  int i = blockIdx.x * blockDim.x + threadIdx.x;
  if (i < n) ws[i] = 0.0f;
}

// ---------------------------------------------------------------------------
// Pass 1: KV[b] += phi_K_chunk^T @ V_chunk  (split-K over s, f32 atomics),
//         k_sum[b] += column sums of phi_K (accumulated in registers).
// Grid: NBATCH * NCHUNK * 4 quadrants; block = 256 threads (8 waves).
// Waves 0-3 stage phi_K rows, waves 4-7 stage V rows (one LDS row per thread:
// 32 strided global loads off one base, 4x ds_store_b128). LDS double-buffered
// so stage(k+1) overlaps the 8 WMMAs of step k; one barrier per K-step.
// ---------------------------------------------------------------------------
__global__ __launch_bounds__(256)
void la_pass1_kv(const float* __restrict__ Kin, const float* __restrict__ Vin,
                 float* __restrict__ KV, float* __restrict__ ksum) {
  __shared__ alignas(16) _Float16 As[2][128][32];  // phi_K^T tiles [d][s]
  __shared__ alignas(16) _Float16 Bs[2][128][32];  // V^T tiles     [e][s]

  const int t   = threadIdx.x;
  const int bid = blockIdx.x;
  const int mi  = (bid >> 1) & 1;                // d-half of KV (A rows)
  const int ni  = bid & 1;                       // e-half of KV (B cols)
  const int c   = (bid >> 2) & (NCHUNK - 1);     // split-K chunk
  const int b   = bid >> 6;                      // batch (4*NCHUNK = 64/batch)
  const int d0  = mi * 128, e0 = ni * 128;
  const int sbase = c * SCHUNK;

  const bool isK = (t < 128);                    // wave-uniform role split
  const int  rl  = t & 127;                      // owned LDS row
  const float* srcBase =
      (isK ? Kin + (size_t)(b * SLEN + sbase) * DFEAT + d0 + rl
           : Vin + (size_t)(b * SLEN + sbase) * DFEAT + e0 + rl);

  float ksumAcc = 0.0f;

  auto stage = [&](int buf, int kk) {
    const float* p = srcBase + (size_t)kk * DFEAT;
    if (kk + 64 < SCHUNK) __builtin_prefetch(p + 64 * DFEAT, 0, 1);
    float v[32];
#pragma unroll
    for (int sl = 0; sl < 32; ++sl) v[sl] = p[(size_t)sl * DFEAT];
    _Float16* dst = isK ? &As[buf][rl][0] : &Bs[buf][rl][0];
#pragma unroll
    for (int j = 0; j < 4; ++j) {
      union { _Float16 h[8]; v4u u; } pk;
#pragma unroll
      for (int q = 0; q < 8; ++q) {
        float x = v[j * 8 + q];
        if (isK) { x = phi_sig(x); ksumAcc += x; }
        pk.h[q] = (_Float16)x;
      }
      ((v4u*)dst)[j] = pk.u;
    }
  };

  const int w  = t >> 5;   // 8 waves
  const int wr = w & 1;    // M group: tiles wr*4 .. wr*4+3
  const int wc = w >> 1;   // N group: tiles wc*2 .. wc*2+1
  v8f acc[4][2] = {};

  stage(0, 0);
  __syncthreads();

  for (int kk = 0; kk < SCHUNK; kk += 32) {
    const int cur = (kk >> 5) & 1;
    if (kk + 32 < SCHUNK) stage(cur ^ 1, kk + 32);  // overlaps WMMAs below

    v16h af[4], bf[2];
#pragma unroll
    for (int mt = 0; mt < 4; ++mt)
      af[mt] = load_frag16x32(&As[cur][0][0], 32, (wr * 4 + mt) * 16, 0);
#pragma unroll
    for (int nt = 0; nt < 2; ++nt)
      bf[nt] = load_frag16x32(&Bs[cur][0][0], 32, (wc * 2 + nt) * 16, 0);
#pragma unroll
    for (int mt = 0; mt < 4; ++mt)
#pragma unroll
      for (int nt = 0; nt < 2; ++nt)
        acc[mt][nt] = __builtin_amdgcn_wmma_f32_16x16x32_f16(
            false, af[mt], false, bf[nt], (short)0, acc[mt][nt], false, false);
    __syncthreads();
  }

  if (ni == 0 && isK) atomicAdd(&ksum[b * DFEAT + d0 + rl], ksumAcc);

  // Split-K reduction into global KV via f32 atomics.
  const int lane = t & 31;
#pragma unroll
  for (int mt = 0; mt < 4; ++mt)
#pragma unroll
    for (int nt = 0; nt < 2; ++nt)
#pragma unroll
      for (int i = 0; i < 8; ++i) {
        int row = d0 + (wr * 4 + mt) * 16 + i + ((lane < 16) ? 0 : 8);
        int col = e0 + (wc * 2 + nt) * 16 + (lane & 15);
        atomicAdd(&KV[b * DFEAT * DFEAT + row * DFEAT + col], acc[mt][nt][i]);
      }
}

// ---------------------------------------------------------------------------
// Pass 2: out = (phi_Q @ KV) / (phi_Q . k_sum), 64 s-rows x 256 cols/block.
// phi_Q staged once (natural A layout); normalizer dot folded into the same
// conversion loop. KV slices: thread-owned column reads prefetched into
// registers one K-step ahead so global loads overlap the WMMAs.
// ---------------------------------------------------------------------------
__global__ __launch_bounds__(256)
void la_pass2_out(const float* __restrict__ Qin, const float* __restrict__ KV,
                  const float* __restrict__ ksum, float* __restrict__ out) {
  __shared__ alignas(16) _Float16 Aq[64][256];  // phi_Q tile [s][d], 32 KB
  __shared__ alignas(16) _Float16 Bk[256][32];  // KV^T slice [e][k], 16 KB
  __shared__ float bot4[64][4];
  __shared__ float bottom[64];

  const int t  = threadIdx.x;
  const int b  = blockIdx.x >> 6;   // 64 row-blocks per batch
  const int s0 = (blockIdx.x & 63) * 64;

  // Stage phi_Q: thread t owns row r = t>>2, cols [qd, qd+64); fold the
  // normalizer partial dot with k_sum into the same pass.
  {
    const int r  = t >> 2;
    const int qd = (t & 3) * 64;
    const float* p  = Qin + (size_t)(b * SLEN + s0 + r) * DFEAT + qd;
    const float* ks = ksum + b * DFEAT + qd;
    float dotp = 0.0f;
#pragma unroll
    for (int j = 0; j < 8; ++j) {
      union { _Float16 h[8]; v4u u; } pk;
#pragma unroll
      for (int q = 0; q < 8; ++q) {
        float ph = phi_sig(p[j * 8 + q]);
        dotp += ph * ks[j * 8 + q];
        pk.h[q] = (_Float16)ph;
      }
      ((v4u*)&Aq[r][qd])[j] = pk.u;
    }
    bot4[r][t & 3] = dotp;
  }
  __syncthreads();
  if (t < 64) bottom[t] = bot4[t][0] + bot4[t][1] + bot4[t][2] + bot4[t][3];

  // Register prefetch of KV slice 0: thread t owns output column e = t.
  const float* kvcol = KV + (size_t)b * DFEAT * DFEAT + t;
  float r[32];
#pragma unroll
  for (int i = 0; i < 32; ++i) r[i] = kvcol[(size_t)i * DFEAT];

  const int w  = t >> 5;
  const int wm = w >> 1;   // M tile 0..3 (16 rows each)
  const int wn = w & 1;    // N half: tiles wn*8 .. wn*8+7
  v8f acc[8] = {};

  for (int k0 = 0; k0 < DFEAT; k0 += 32) {
    __syncthreads();                       // previous Bk consumers done
#pragma unroll
    for (int j = 0; j < 4; ++j) {
      union { _Float16 h[8]; v4u u; } pk;
#pragma unroll
      for (int q = 0; q < 8; ++q) pk.h[q] = (_Float16)r[j * 8 + q];
      ((v4u*)&Bk[t][0])[j] = pk.u;
    }
    __syncthreads();                       // Bk published
    if (k0 + 32 < DFEAT) {                 // loads fly during the WMMAs below
      const float* pn = kvcol + (size_t)(k0 + 32) * DFEAT;
#pragma unroll
      for (int i = 0; i < 32; ++i) r[i] = pn[(size_t)i * DFEAT];
    }

    v16h a = load_frag16x32(&Aq[0][0], 256, wm * 16, k0);
#pragma unroll
    for (int nt = 0; nt < 8; ++nt) {
      v16h bfrag = load_frag16x32(&Bk[0][0], 32, (wn * 8 + nt) * 16, 0);
      acc[nt] = __builtin_amdgcn_wmma_f32_16x16x32_f16(
          false, a, false, bfrag, (short)0, acc[nt], false, false);
    }
  }

  // Divide by normalizer and store (C/D layout: VGPR i -> M=m0+i / m0+8+i).
  const int lane = t & 31;
#pragma unroll
  for (int nt = 0; nt < 8; ++nt)
#pragma unroll
    for (int i = 0; i < 8; ++i) {
      int row = wm * 16 + i + ((lane < 16) ? 0 : 8);
      int col = (wn * 8 + nt) * 16 + (lane & 15);
      out[(size_t)(b * SLEN + s0 + row) * DFEAT + col] = acc[nt][i] / bottom[row];
    }
}

extern "C" void kernel_launch(void* const* d_in, const int* in_sizes, int n_in,
                              void* d_out, int out_size, void* d_ws, size_t ws_size,
                              hipStream_t stream) {
  (void)in_sizes; (void)n_in; (void)out_size; (void)ws_size;
  const float* Q = (const float*)d_in[0];
  const float* K = (const float*)d_in[1];
  const float* V = (const float*)d_in[2];
  float* out  = (float*)d_out;
  float* KV   = (float*)d_ws;                        // 8 * 256*256 f32 = 2 MB
  float* ksum = KV + NBATCH * DFEAT * DFEAT;         // 8 * 256 f32

  const int nws = NBATCH * DFEAT * DFEAT + NBATCH * DFEAT;
  la_zero_ws<<<(nws + 255) / 256, 256, 0, stream>>>((float*)d_ws, nws);
  la_pass1_kv<<<NBATCH * NCHUNK * 4, 256, 0, stream>>>(K, V, KV, ksum);
  la_pass2_out<<<NBATCH * (SLEN / 64), 256, 0, stream>>>(Q, KV, ksum, out);
}
